// FoundationalTimeSeriesModel_10797547782902
// MI455X (gfx1250) — matrix operand order repaired and checked by reading, not compiled, not run on hardware
//
// MI455X (gfx1250, CDNA5) implementation of FoundationalTimeSeriesModel.
// Centerpiece: LDS-resident TCN megakernel (8 dilated convs as implicit GEMMs
// via v_wmma_f32_16x16x32_f16, fused LN + masked temporal pooling), with
// async global->LDS staging of the input column (global_load_async_to_lds_b32
// + s_wait_asynccnt). All token-level matmuls use a generic direct-WMMA GEMM.
// wave32 throughout.

#include <hip/hip_runtime.h>
#include <hip/hip_bf16.h>
#include <math.h>

typedef __attribute__((ext_vector_type(16))) _Float16 v16h;
typedef __attribute__((ext_vector_type(8)))  _Float16 v8h;
typedef __attribute__((ext_vector_type(8)))  float    v8f;

#define TS_S    512
#define TS_DT   64
#define TS_PADL 16                 // (K-1)*max_dil = 2*8
#define TS_SROW (TS_S + TS_PADL)   // 528

// ---------------------------------------------------------------------------
// TCN megakernel: one block per (b,n) sequence. 256 threads = 8 waves.
// LDS: hbuf + obuf (528x64 f16 each, time-major) + x stage + LN stats ~139 KB.
// Each conv: out[64,512] = W[64,192] x im2col[192,512], WMMA 16x16x32 f16.
// Wave w: mtile = w&3 (c_out block), ntiles (w>>2)*16 .. +15 (time blocks).
// ---------------------------------------------------------------------------
__global__ __launch_bounds__(256) void tcn_mega(
    const float* __restrict__ x,       // [B,S,N]
    const float* __restrict__ mask,    // [B,N]
    const float* __restrict__ proj_w,  // [1,64]
    const float* __restrict__ proj_b,  // [64]
    const float* __restrict__ w1,      // [4,64,64,3]
    const float* __restrict__ b1,      // [4,64]
    const float* __restrict__ w2,      // [4,64,64,3]
    const float* __restrict__ b2,      // [4,64]
    const float* __restrict__ lng,     // [64]
    const float* __restrict__ lnb,     // [64]
    float* __restrict__ pooled,        // [1024,64]
    float* __restrict__ stlast)        // [1024,64]
{
  __shared__ __attribute__((aligned(16))) _Float16 hbuf[TS_SROW * TS_DT];
  __shared__ __attribute__((aligned(16))) _Float16 obuf[TS_SROW * TS_DT];
  __shared__ __attribute__((aligned(16))) float xstage[TS_S];
  __shared__ float smu[TS_S];
  __shared__ float srstd[TS_S];
  __shared__ float partial[4 * TS_DT];

  const int bn  = blockIdx.x;
  const int b   = bn >> 5;
  const int n   = bn & 31;
  const int tid = threadIdx.x;

  // ---- async-stage this sequence's x column (stride-128B gather) to LDS ---
  // global_load_async_to_lds_b32: vdst = LDS byte address, vaddr = 64-bit
  // global address, tracked by ASYNCcnt (no VGPR round-trip).
  {
    for (int t = tid; t < TS_S; t += 256) {
      unsigned lds_addr = (unsigned)(size_t)&xstage[t];
      const float* gptr = &x[(b * TS_S + t) * 32 + n];
      asm volatile("global_load_async_to_lds_b32 %0, %1, off"
                   :
                   : "v"(lds_addr), "v"(gptr)
                   : "memory");
    }
    asm volatile("s_wait_asynccnt 0x0" ::: "memory");
  }
  __syncthreads();

  // ---- init: h[t][c] = x*proj_w + proj_b + PE ; zero pad rows (t<0) ----
  for (int idx = tid; idx < TS_SROW * TS_DT; idx += 256) {
    int row = idx >> 6, c = idx & 63;
    float v = 0.f;
    if (row >= TS_PADL) {
      int t = row - TS_PADL;
      float xi   = xstage[t];
      float freq = __expf(-(float)(c & ~1) * (9.210340371976184f / 64.f));
      float ang  = (float)t * freq;
      float pe   = (c & 1) ? __cosf(ang) : __sinf(ang);
      v = xi * proj_w[c] + proj_b[c] + pe;
    }
    hbuf[idx] = (_Float16)v;
    if (row < TS_PADL) obuf[idx] = (_Float16)0.f;
  }
  __syncthreads();

  const int lane  = tid & 31;
  const int wv    = tid >> 5;
  const int mtile = wv & 3;
  const int c0    = mtile * 16;
  const int ngrp  = wv >> 2;         // 0..1
  const int lm    = lane & 15;
  const int lhi   = lane >> 4;       // 0 or 1

  for (int layer = 0; layer < 4; ++layer) {
    const int dil = 1 << layer;
    for (int hf = 0; hf < 2; ++hf) {
      const float* wg = (hf ? w2 : w1) + layer * TS_DT * TS_DT * 3;
      const float* bg = (hf ? b2 : b1) + layer * TS_DT;
      _Float16* src = hf ? obuf : hbuf;
      _Float16* dst = hf ? hbuf : obuf;

      // ---- preload 6 A fragments (weights) from global; reused 16x ----
      // A layout (16-bit A 16x32): lanes<16 K={0..7,16..23}; lanes>=16 +8.
      v16h A[6];
#pragma unroll
      for (int kt = 0; kt < 6; ++kt) {
        int k = kt >> 1, cb = (kt & 1) * 32;
        const float* wr = wg + (c0 + lm) * 192 + k;  // [c_out][c_in][k]
        v16h a;
#pragma unroll
        for (int i = 0; i < 8; ++i)
          a[i] = (_Float16)wr[(cb + lhi * 8 + i) * 3];
#pragma unroll
        for (int i = 0; i < 8; ++i)
          a[8 + i] = (_Float16)wr[(cb + 16 + lhi * 8 + i) * 3];
        A[kt] = a;
      }

      for (int j = 0; j < 16; ++j) {
        const int t0 = (ngrp * 16 + j) * 16;
        v8f acc = {};
#pragma unroll
        for (int kt = 0; kt < 6; ++kt) {
          int k  = kt >> 1;
          int cb = (kt & 1) * 32;
          int row = t0 + lm + TS_PADL - k * dil;      // time-shifted im2col
          // B layout (32x16): lanes<16 K={0..7,16..23}; lanes>=16 +8,
          // -> two ds_load_b128 per lane.
          const v8h* p = (const v8h*)&src[row * TS_DT + cb + lhi * 8];
          v8h blo = p[0], bhi = p[2];
          v16h bfrag = __builtin_shufflevector(
              blo, bhi, 0, 1, 2, 3, 4, 5, 6, 7, 8, 9, 10, 11, 12, 13, 14, 15);
          acc = __builtin_amdgcn_wmma_f32_16x16x32_f16(
              false, A[kt], false, bfrag, (short)0, acc, false, false);
        }
        // ---- epilogue: bias (+residual) + relu, pack 8xf16, one b128 store
        const int t   = t0 + lm;
        const int chi = c0 + lhi * 8;
        v8h rv;
        if (hf) rv = *(const v8h*)&dst[(t + TS_PADL) * TS_DT + chi];
        v8h dvec;
#pragma unroll
        for (int g = 0; g < 8; ++g) {
          float v = acc[g] + bg[chi + g];
          if (hf) v += (float)rv[g];
          v = fmaxf(v, 0.f);
          dvec[g] = (_Float16)v;
        }
        *(v8h*)&dst[(t + TS_PADL) * TS_DT + chi] = dvec;
      }
      __syncthreads();
    }
  }

  // ---- fused LayerNorm(DT) + masked mean over S + st_last ----
  for (int t = tid; t < TS_S; t += 256) {
    float s = 0.f, ss = 0.f;
#pragma unroll 8
    for (int c = 0; c < 64; ++c) {
      float v = (float)hbuf[(t + TS_PADL) * TS_DT + c];
      s += v; ss += v * v;
    }
    float mu  = s * (1.f / 64.f);
    float var = ss * (1.f / 64.f) - mu * mu;
    smu[t]   = mu;
    srstd[t] = rsqrtf(var + 1e-5f);
  }
  __syncthreads();
  {
    int c = tid & 63, grp = tid >> 6;
    float acc2 = 0.f;
    for (int t = grp; t < TS_S; t += 4)
      acc2 += ((float)hbuf[(t + TS_PADL) * TS_DT + c] - smu[t]) * srstd[t];
    partial[grp * 64 + c] = acc2;
  }
  __syncthreads();
  if (tid < 64) {
    float m = mask[bn];
    float s = partial[tid] + partial[64 + tid] + partial[128 + tid] +
              partial[192 + tid];
    pooled[bn * 64 + tid] = m * ((s * (1.f / 512.f)) * lng[tid] + lnb[tid]);
    float v = (float)hbuf[(511 + TS_PADL) * TS_DT + tid];
    stlast[bn * 64 + tid] =
        m * (((v - smu[511]) * srstd[511]) * lng[tid] + lnb[tid]);
  }
}

// ---------------------------------------------------------------------------
// Generic direct-WMMA GEMM: out[M,Nn] = op(A[M,Kk] @ W[Kk,Nn] + bias (+pos)
// (+res)).  f32 global -> f16 fragments in registers.  One 16x16 tile/wave.
// M,Nn multiples of 16; Kk multiple of 32.
// ---------------------------------------------------------------------------
__global__ __launch_bounds__(128) void gemm_wmma(
    const float* __restrict__ A, const float* __restrict__ W,
    const float* __restrict__ bias, const float* __restrict__ res,
    const float* __restrict__ posb, float* __restrict__ out,
    int M, int Nn, int Kk, int relu)
{
  const int lane   = threadIdx.x & 31;
  const int wv     = threadIdx.x >> 5;
  const int tile   = blockIdx.x * 4 + wv;
  const int ntiles = Nn >> 4;
  const int total  = (M >> 4) * ntiles;
  if (tile >= total) return;
  const int mt  = tile / ntiles;
  const int nt  = tile % ntiles;
  const int lm  = lane & 15;
  const int lhi = lane >> 4;
  const int col = nt * 16 + lm;

  v8f acc = {};
  for (int kb = 0; kb < Kk; kb += 32) {
    const float* ar = A + (mt * 16 + lm) * Kk + kb;
    __builtin_prefetch(ar + 32, 0, 0);   // global_prefetch for next k-slice
    v16h a;
#pragma unroll
    for (int i = 0; i < 8; ++i) a[i] = (_Float16)ar[lhi * 8 + i];
#pragma unroll
    for (int i = 0; i < 8; ++i) a[8 + i] = (_Float16)ar[16 + lhi * 8 + i];
    const float* br = W + (kb + lhi * 8) * Nn + col;
    v16h bv;
#pragma unroll
    for (int i = 0; i < 16; ++i) bv[i] = (_Float16)br[(((i & 8) << 1) + (i & 7)) * Nn];
    acc = __builtin_amdgcn_wmma_f32_16x16x32_f16(
        false, a, false, bv, (short)0, acc, false, false);
  }
  const int row0 = mt * 16 + lhi * 8;
#pragma unroll
  for (int g = 0; g < 8; ++g) {
    int r = row0 + g;
    float v = acc[g];
    if (bias) v += bias[col];
    if (posb) v += posb[(r & 31) * Nn + col];
    if (res)  v += res[r * Nn + col];
    if (relu) v = fmaxf(v, 0.f);
    out[r * Nn + col] = v;
  }
}

// ---------------------------------------------------------------------------
// Attention: one wave per (b, head). 32 queries x 32 keys, hd=16, masked.
// ---------------------------------------------------------------------------
__global__ __launch_bounds__(32) void attn_kernel(
    const float* __restrict__ qkv, const float* __restrict__ mask,
    float* __restrict__ outp)
{
  const int bh = blockIdx.x;
  const int b  = bh >> 3;
  const int h  = bh & 7;
  const int i  = threadIdx.x;  // query index

  float q[16];
#pragma unroll
  for (int d = 0; d < 16; ++d) q[d] = qkv[(b * 32 + i) * 384 + h * 16 + d];

  float sc[32];
#pragma unroll
  for (int j = 0; j < 32; ++j) {
    const float* kr = &qkv[(b * 32 + j) * 384 + 128 + h * 16];
    float s = 0.f;
#pragma unroll
    for (int d = 0; d < 16; ++d) s += q[d] * kr[d];
    s *= 0.25f;                                  // 1/sqrt(16)
    if (mask[b * 32 + j] == 0.f) s = -1e9f;
    sc[j] = s;
  }
  float mx = sc[0];
#pragma unroll
  for (int j = 1; j < 32; ++j) mx = fmaxf(mx, sc[j]);
  float den = 0.f;
#pragma unroll
  for (int j = 0; j < 32; ++j) { sc[j] = __expf(sc[j] - mx); den += sc[j]; }
  const float inv = 1.f / den;
  float o[16];
#pragma unroll
  for (int d = 0; d < 16; ++d) o[d] = 0.f;
#pragma unroll
  for (int j = 0; j < 32; ++j) {
    const float* vr = &qkv[(b * 32 + j) * 384 + 256 + h * 16];
    float pp = sc[j] * inv;
#pragma unroll
    for (int d = 0; d < 16; ++d) o[d] += pp * vr[d];
  }
#pragma unroll
  for (int d = 0; d < 16; ++d)
    outp[(b * 32 + i) * 128 + h * 16 + d] = o[d];
}

// ---------------------------------------------------------------------------
// LayerNorm over D=128, one block per row; optional per-row mask scale.
// ---------------------------------------------------------------------------
__global__ __launch_bounds__(128) void ln128_kernel(
    const float* __restrict__ in, float* __restrict__ out,
    const float* __restrict__ g, const float* __restrict__ bta,
    const float* __restrict__ mask)
{
  __shared__ float red[128];
  const int r = blockIdx.x, c = threadIdx.x;
  float v = in[r * 128 + c];
  red[c] = v; __syncthreads();
  for (int s2 = 64; s2 > 0; s2 >>= 1) {
    if (c < s2) red[c] += red[c + s2];
    __syncthreads();
  }
  float mean = red[0] * (1.f / 128.f);
  __syncthreads();
  float d = v - mean;
  red[c] = d * d; __syncthreads();
  for (int s2 = 64; s2 > 0; s2 >>= 1) {
    if (c < s2) red[c] += red[c + s2];
    __syncthreads();
  }
  float var = red[0] * (1.f / 128.f);
  float o = d * rsqrtf(var + 1e-5f) * g[c] + bta[c];
  if (mask) o *= mask[r];
  out[r * 128 + c] = o;
}

// ---------------------------------------------------------------------------
// Router stats: masked mean/std over N per batch -> mean_ctx, router_in.
// ---------------------------------------------------------------------------
__global__ __launch_bounds__(128) void router_kernel(
    const float* __restrict__ ctx, const float* __restrict__ mask,
    float* __restrict__ meanctx, float* __restrict__ routerin)
{
  const int b = blockIdx.x, c = threadIdx.x;
  float cnt = 0.f;
  for (int n2 = 0; n2 < 32; ++n2) cnt += mask[b * 32 + n2];
  cnt = fmaxf(cnt, 1.f);
  float s = 0.f, ss = 0.f;
  for (int n2 = 0; n2 < 32; ++n2) {
    float v = ctx[(b * 32 + n2) * 128 + c];
    s += v; ss += v * v;
  }
  float mean = s / cnt;
  float var  = ss / cnt - mean * mean;
  meanctx[b * 128 + c] = mean;
  routerin[b * 256 + c] = mean;
  routerin[b * 256 + 128 + c] = sqrtf(fmaxf(var, 1e-6f));
}

// ---------------------------------------------------------------------------
// MoE top-2 combine from dense per-expert outputs.
// ---------------------------------------------------------------------------
__global__ __launch_bounds__(128) void moe_combine_kernel(
    const float* __restrict__ logits, const float* __restrict__ allo,
    float* __restrict__ out, int expStride, const float* __restrict__ mask)
{
  const int row = blockIdx.x, c = threadIdx.x;
  float L[16];
#pragma unroll
  for (int e = 0; e < 16; ++e) L[e] = logits[row * 16 + e];
  int i0 = 0; float v0 = L[0];
#pragma unroll
  for (int e = 1; e < 16; ++e) if (L[e] > v0) { v0 = L[e]; i0 = e; }
  int i1 = 0; float v1 = -INFINITY;
#pragma unroll
  for (int e = 0; e < 16; ++e)
    if (e != i0 && L[e] > v1) { v1 = L[e]; i1 = e; }
  float e1 = __expf(v1 - v0);
  float w0 = 1.f / (1.f + e1), w1 = e1 / (1.f + e1);
  float o = w0 * allo[i0 * expStride + row * 128 + c] +
            w1 * allo[i1 * expStride + row * 128 + c];
  if (mask) o *= mask[row];
  out[row * 128 + c] = o;
}

// ---------------------------------------------------------------------------
// Aux load-balancing loss from logits_f (32 rows x 16 experts).
// ---------------------------------------------------------------------------
__global__ __launch_bounds__(32) void aux_kernel(
    const float* __restrict__ logits, float* __restrict__ outp)
{
  __shared__ float psum[16];
  __shared__ float pcnt[16];
  const int t = threadIdx.x;   // row index 0..31
  if (t < 16) { psum[t] = 0.f; pcnt[t] = 0.f; }
  __syncthreads();
  float L[16];
#pragma unroll
  for (int e = 0; e < 16; ++e) L[e] = logits[t * 16 + e];
  float mx = L[0];
#pragma unroll
  for (int e = 1; e < 16; ++e) mx = fmaxf(mx, L[e]);
  float den = 0.f, ex[16];
#pragma unroll
  for (int e = 0; e < 16; ++e) { ex[e] = __expf(L[e] - mx); den += ex[e]; }
#pragma unroll
  for (int e = 0; e < 16; ++e) atomicAdd(&psum[e], ex[e] / den);
  int i0 = 0; float v0 = L[0];
#pragma unroll
  for (int e = 1; e < 16; ++e) if (L[e] > v0) { v0 = L[e]; i0 = e; }
  int i1 = 0; float v1 = -INFINITY;
#pragma unroll
  for (int e = 0; e < 16; ++e)
    if (e != i0 && L[e] > v1) { v1 = L[e]; i1 = e; }
  atomicAdd(&pcnt[i0], 1.f);
  atomicAdd(&pcnt[i1], 1.f);
  __syncthreads();
  if (t == 0) {
    float loss = 0.f;
    for (int e = 0; e < 16; ++e)
      loss += (psum[e] / 32.f) * (pcnt[e] / 64.f);
    outp[0] = 16.f * loss;
  }
}

// ---------------------------------------------------------------------------
// Output heads.
// ---------------------------------------------------------------------------
__global__ __launch_bounds__(256) void pred_head(
    const float* __restrict__ st, const float* __restrict__ ctx,
    const float* __restrict__ moef, const float* __restrict__ w,
    const float* __restrict__ bia, float* __restrict__ outp)
{
  int i = blockIdx.x * 256 + threadIdx.x;
  if (i >= 3072) return;
  int row = i / 3, cc = i % 3, b = row >> 5;
  float acc = bia[cc];
  for (int j = 0; j < 64; ++j)  acc += st[row * 64 + j]   * w[j * 3 + cc];
  for (int j = 0; j < 128; ++j) acc += ctx[row * 128 + j] * w[(64 + j) * 3 + cc];
  for (int j = 0; j < 128; ++j) acc += moef[b * 128 + j]  * w[(192 + j) * 3 + cc];
  outp[i] = acc;
}

__global__ __launch_bounds__(96) void fail_head(
    const float* __restrict__ moefail, const float* __restrict__ w,
    const float* __restrict__ bia, float* __restrict__ outp)
{
  int i = threadIdx.x;
  int b = i / 3, cc = i % 3;
  float acc = bia[cc];
  for (int j = 0; j < 128; ++j) acc += moefail[b * 128 + j] * w[j * 3 + cc];
  outp[i] = acc;
}

__global__ __launch_bounds__(256) void rca_head(
    const float* __restrict__ st, const float* __restrict__ ctx,
    const float* __restrict__ moer, const float* __restrict__ w,
    const float* __restrict__ bia, float* __restrict__ outp)
{
  int row = blockIdx.x * 256 + threadIdx.x;
  if (row >= 1024) return;
  float acc = bia[0];
  for (int j = 0; j < 64; ++j)  acc += st[row * 64 + j]    * w[j];
  for (int j = 0; j < 128; ++j) acc += ctx[row * 128 + j]  * w[64 + j];
  for (int j = 0; j < 128; ++j) acc += moer[row * 128 + j] * w[192 + j];
  outp[row] = acc;
}

// ---------------------------------------------------------------------------
// Host orchestration.
// ---------------------------------------------------------------------------
extern "C" void kernel_launch(void* const* d_in, const int* in_sizes, int n_in,
                              void* d_out, int out_size, void* d_ws,
                              size_t ws_size, hipStream_t stream)
{
  const float* x          = (const float*)d_in[0];
  const float* mask       = (const float*)d_in[1];
  const float* proj_w     = (const float*)d_in[2];
  const float* proj_b     = (const float*)d_in[3];
  const float* tcn_w1     = (const float*)d_in[4];
  const float* tcn_b1     = (const float*)d_in[5];
  const float* tcn_w2     = (const float*)d_in[6];
  const float* tcn_b2     = (const float*)d_in[7];
  const float* tcn_ln_g   = (const float*)d_in[8];
  const float* tcn_ln_b   = (const float*)d_in[9];
  const float* pool_w     = (const float*)d_in[10];
  const float* pool_b     = (const float*)d_in[11];
  const float* pos_inter  = (const float*)d_in[12];
  const float* tx_ln1_g   = (const float*)d_in[13];
  const float* tx_ln1_b   = (const float*)d_in[14];
  const float* tx_qkv_w   = (const float*)d_in[15];
  const float* tx_qkv_b   = (const float*)d_in[16];
  const float* tx_out_w   = (const float*)d_in[17];
  const float* tx_out_b   = (const float*)d_in[18];
  const float* tx_ln2_g   = (const float*)d_in[19];
  const float* tx_ln2_b   = (const float*)d_in[20];
  const float* tx_ff1_w   = (const float*)d_in[21];
  const float* tx_ff1_b   = (const float*)d_in[22];
  const float* tx_ff2_w   = (const float*)d_in[23];
  const float* tx_ff2_b   = (const float*)d_in[24];
  const float* tx_fln_g   = (const float*)d_in[25];
  const float* tx_fln_b   = (const float*)d_in[26];
  const float* exp_w1     = (const float*)d_in[27];
  const float* exp_b1     = (const float*)d_in[28];
  const float* exp_w2     = (const float*)d_in[29];
  const float* exp_b2     = (const float*)d_in[30];
  const float* gate_f_w   = (const float*)d_in[31];
  const float* gate_f_b   = (const float*)d_in[32];
  const float* gate_fl_w  = (const float*)d_in[33];
  const float* gate_fl_b  = (const float*)d_in[34];
  const float* gate_rc_w  = (const float*)d_in[35];
  const float* gate_rc_b  = (const float*)d_in[36];
  const float* pred_w     = (const float*)d_in[37];
  const float* pred_b     = (const float*)d_in[38];
  const float* fail_w     = (const float*)d_in[39];
  const float* fail_b     = (const float*)d_in[40];
  const float* rca_w      = (const float*)d_in[41];
  const float* rca_b      = (const float*)d_in[42];
  float* out = (float*)d_out;

  // workspace layout (floats)
  float* p = (float*)d_ws;
  float* pooled   = p; p += 1024 * 64;
  float* stlast   = p; p += 1024 * 64;
  float* z        = p; p += 1024 * 128;
  float* ybuf     = p; p += 1024 * 128;
  float* qkvbuf   = p; p += 1024 * 384;
  float* attnout  = p; p += 1024 * 128;
  float* ffbuf    = p; p += 1024 * 256;
  float* ctx      = p; p += 1024 * 128;
  float* meanctx  = p; p += 32 * 128;
  float* routerin = p; p += 32 * 256;
  float* lg_f     = p; p += 32 * 16;
  float* lg_fl    = p; p += 32 * 16;
  float* lg_rc    = p; p += 1024 * 16;
  float* h1buf    = p; p += 1024 * 256;
  float* h1m      = p; p += 32 * 256;
  float* allo_tok = p; p += 16 * 1024 * 128;
  float* allo_m   = p; p += 16 * 32 * 128;
  float* moe_f    = p; p += 32 * 128;
  float* moe_fl   = p; p += 32 * 128;
  float* moe_rc   = p; p += 1024 * 128;

  auto G = [&](const float* A, const float* W, const float* bias,
               const float* res, const float* posb, float* o,
               int M, int Nn, int Kk, int relu) {
    int tiles = (M / 16) * (Nn / 16);
    gemm_wmma<<<(tiles + 3) / 4, 128, 0, stream>>>(A, W, bias, res, posb, o,
                                                   M, Nn, Kk, relu);
  };

  // 1) TCN megakernel: proj+PE -> 8 dilated convs -> LN -> masked pool
  tcn_mega<<<1024, 256, 0, stream>>>(x, mask, proj_w, proj_b, tcn_w1, tcn_b1,
                                     tcn_w2, tcn_b2, tcn_ln_g, tcn_ln_b,
                                     pooled, stlast);

  // 2) pooled @ pool_w + pool_b + pos_inter[n]
  G(pooled, pool_w, pool_b, nullptr, pos_inter, z, 1024, 128, 64, 0);

  // 3) transformer, 2 layers
  for (int l = 0; l < 2; ++l) {
    ln128_kernel<<<1024, 128, 0, stream>>>(z, ybuf, tx_ln1_g + l * 128,
                                           tx_ln1_b + l * 128, nullptr);
    G(ybuf, tx_qkv_w + l * 128 * 384, tx_qkv_b + l * 384, nullptr, nullptr,
      qkvbuf, 1024, 384, 128, 0);
    attn_kernel<<<256, 32, 0, stream>>>(qkvbuf, mask, attnout);
    G(attnout, tx_out_w + l * 128 * 128, tx_out_b + l * 128, z, nullptr, z,
      1024, 128, 128, 0);
    ln128_kernel<<<1024, 128, 0, stream>>>(z, ybuf, tx_ln2_g + l * 128,
                                           tx_ln2_b + l * 128, nullptr);
    G(ybuf, tx_ff1_w + l * 128 * 256, tx_ff1_b + l * 256, nullptr, nullptr,
      ffbuf, 1024, 256, 128, 1);
    G(ffbuf, tx_ff2_w + l * 256 * 128, tx_ff2_b + l * 128, z, nullptr, z,
      1024, 128, 256, 0);
  }

  // 4) final LN (masked) + router stats
  ln128_kernel<<<1024, 128, 0, stream>>>(z, ctx, tx_fln_g, tx_fln_b, mask);
  router_kernel<<<32, 128, 0, stream>>>(ctx, mask, meanctx, routerin);

  // 5) gate logits (WMMA, N=16)
  G(routerin, gate_f_w,  gate_f_b,  nullptr, nullptr, lg_f,  32,   16, 256, 0);
  G(routerin, gate_fl_w, gate_fl_b, nullptr, nullptr, lg_fl, 32,   16, 256, 0);
  G(ctx,      gate_rc_w, gate_rc_b, nullptr, nullptr, lg_rc, 1024, 16, 128, 0);

  // 6) dense expert evaluation (token set + mean_ctx set)
  for (int e = 0; e < 16; ++e) {
    const float* w1e = exp_w1 + e * 128 * 256;
    const float* b1e = exp_b1 + e * 256;
    const float* w2e = exp_w2 + e * 256 * 128;
    const float* b2e = exp_b2 + e * 128;
    G(ctx,   w1e, b1e, nullptr, nullptr, h1buf, 1024, 256, 128, 1);
    G(h1buf, w2e, b2e, nullptr, nullptr, allo_tok + e * 1024 * 128,
      1024, 128, 256, 0);
    G(meanctx, w1e, b1e, nullptr, nullptr, h1m, 32, 256, 128, 1);
    G(h1m, w2e, b2e, nullptr, nullptr, allo_m + e * 32 * 128,
      32, 128, 256, 0);
  }

  // 7) top-2 combine + aux loss
  moe_combine_kernel<<<32, 128, 0, stream>>>(lg_f,  allo_m,   moe_f,
                                             32 * 128, nullptr);
  moe_combine_kernel<<<32, 128, 0, stream>>>(lg_fl, allo_m,   moe_fl,
                                             32 * 128, nullptr);
  moe_combine_kernel<<<1024, 128, 0, stream>>>(lg_rc, allo_tok, moe_rc,
                                               1024 * 128, mask);
  aux_kernel<<<1, 32, 0, stream>>>(lg_f, out + 4192);

  // 8) heads -> d_out: pred[3072] | fail[96] | rca[1024] | aux[1]
  pred_head<<<12, 256, 0, stream>>>(stlast, ctx, moe_f, pred_w, pred_b, out);
  fail_head<<<1, 96, 0, stream>>>(moe_fl, fail_w, fail_b, out + 3072);
  rca_head<<<4, 256, 0, stream>>>(stlast, ctx, moe_rc, rca_w, rca_b,
                                  out + 3168);
}
